// SuperGlueMultiHeadAttention_29927332119004
// MI455X (gfx1250) — matrix operand order repaired and checked by reading, not compiled, not run on hardware
//
#include <hip/hip_runtime.h>

typedef __attribute__((ext_vector_type(16))) _Float16 v16h;
typedef __attribute__((ext_vector_type(8)))  _Float16 v8h;
typedef __attribute__((ext_vector_type(8)))  float    v8f;

typedef __attribute__((ext_vector_type(4))) unsigned int u32x4;
typedef __attribute__((ext_vector_type(8))) int          i32x8;
typedef __attribute__((ext_vector_type(4))) int          i32x4;

#define FEAT  256
#define SEQ   2048
#define BATCH 4
#define NHEAD 4
#define HDIM  64

#if __has_builtin(__builtin_amdgcn_tensor_load_to_lds) && \
    __has_builtin(__builtin_amdgcn_s_wait_tensorcnt)
#define USE_TDM 1
#else
#define USE_TDM 0
#endif

__device__ __forceinline__ v8f wmma_f16(v16h a, v16h b, v8f c) {
    return __builtin_amdgcn_wmma_f32_16x16x32_f16(
        false, a, false, b, (short)0, c, false, false);
}

__device__ __forceinline__ v16h cat8(v8h lo, v8h hi) {
    return __builtin_shufflevector(lo, hi, 0, 1, 2, 3, 4, 5, 6, 7,
                                           8, 9, 10, 11, 12, 13, 14, 15);
}

// SWAPX16: exchange with lane^16 (group-of-32 ds_swizzle, xor_mask=0x10, and=0x1F)
__device__ __forceinline__ float swap16(float v) {
    return __int_as_float(__builtin_amdgcn_ds_swizzle(__float_as_int(v), 0x401F));
}

// generic LDS pointer -> 32-bit LDS byte address (low 32 bits of generic addr)
__device__ __forceinline__ unsigned lds_addr32(const void* p) {
    return (unsigned)(size_t)p;
}

#if USE_TDM
// TDM 2D tile load: nrows rows of row_qw 8-byte qwords, global row stride
// stride_qw qwords, packed contiguously into LDS at lds_addr.
// 6-arg builtin form: (g0, g1, g2, g3, g4, cpol) -- groups 2..4 unused here.
__device__ __forceinline__ void tdm_load_2d(const void* gsrc, unsigned lds_addr,
                                            unsigned row_qw, unsigned nrows,
                                            unsigned stride_qw)
{
    const unsigned long long ga = (unsigned long long)(size_t)gsrc;
    u32x4 g0;
    g0[0] = 1u;                                   // count=1 (valid), user mode
    g0[1] = lds_addr;                             // D#.lds_addr
    g0[2] = (unsigned)ga;                         // global_addr[31:0]
    g0[3] = (unsigned)(ga >> 32) | (2u << 30);    // global_addr[56:32] | type=2
    i32x8 g1;
    g1[0] = (int)(3u << 16);                      // wg_mask=0, data_size=3 (8B)
    g1[1] = (int)((stride_qw & 0xFFFFu) << 16);   // tensor_dim0[15:0]
    g1[2] = (int)((stride_qw >> 16) | ((nrows & 0xFFFFu) << 16)); // dim0 hi | tensor_dim1 lo
    g1[3] = (int)((nrows >> 16) | (row_qw << 16));                // dim1 hi | tile_dim0
    g1[4] = (int)(nrows & 0xFFFFu);               // tile_dim1 | tile_dim2=0
    g1[5] = (int)stride_qw;                       // tensor_dim0_stride[31:0]
    g1[6] = 0;
    g1[7] = 0;
    const i32x4 z4 = {0, 0, 0, 0};
    const i32x8 z8 = {0, 0, 0, 0, 0, 0, 0, 0};
    __builtin_amdgcn_tensor_load_to_lds(g0, g1, z4, z4, z8, 0);
}
#endif

// ---------------------------------------------------------------------------
// GEMM: Y = W(256x256,f32) * X(f32 [B,256,SEQ]) + bias, 8 waves/block,
// each wave one 16x16 tile, K=256 as 8 x v_wmma_f32_16x16x32_f16.
// W tile is staged once per block in LDS as f16 (A-operand -> 2x ds_load_b128).
// MODE 0: Y f16 [B,H,seq,d] | MODE 1: Y f16 [B,H,d,seq] | MODE 2: Y f32 [B,256,SEQ]
// ---------------------------------------------------------------------------
template <int MODE>
__global__ void __launch_bounds__(256)
gemm256(const float* __restrict__ W, const float* __restrict__ bias,
        const float* __restrict__ X, void* __restrict__ Yv)
{
    __shared__ __align__(32) _Float16 aT[16 * FEAT];

    const int lane = threadIdx.x & 31;
    const int wid  = threadIdx.x >> 5;
    const int hi   = lane >> 4;
    const int l16  = lane & 15;

    const int bid   = blockIdx.x;
    const int b     = bid >> 8;
    const int mtile = (bid >> 4) & 15;
    const int ntile = (bid & 15) * 8 + wid;

    const int co0 = mtile * 16;
    const int n0  = ntile * 16;

    {
        const int row = threadIdx.x >> 4;
        const int c0  = (threadIdx.x & 15) * 16;
        const float* src = W + (size_t)(co0 + row) * FEAT + c0;
        v8h h0 = {}, h1 = {};
        #pragma unroll
        for (int i = 0; i < 8; ++i) {
            h0[i] = (_Float16)src[i];
            h1[i] = (_Float16)src[8 + i];
        }
        *(v8h*)(aT + row * FEAT + c0)     = h0;
        *(v8h*)(aT + row * FEAT + c0 + 8) = h1;
    }
    __syncthreads();

    const float* Xcol = X + (size_t)b * FEAT * SEQ + n0 + l16;

    v8f acc = {};
    #pragma unroll 1
    for (int kk = 0; kk < FEAT; kk += 32) {
        v8h alo = *(const v8h*)(aT + l16 * FEAT + kk + 8 * hi);
        v8h ahi = *(const v8h*)(aT + l16 * FEAT + kk + 16 + 8 * hi);
        v16h a  = cat8(alo, ahi);
        v16h bm = {};
        #pragma unroll
        for (int e = 0; e < 16; ++e)
            bm[e] = (_Float16)Xcol[(size_t)(kk + (e & 7) + 8 * hi + 16 * (e >> 3)) * SEQ];
        acc = wmma_f16(a, bm, acc);
    }

    #pragma unroll
    for (int r = 0; r < 8; ++r) {
        const int co = co0 + r + 8 * hi;
        const float v = acc[r] + bias[co];
        const int d = co >> 2, hh = co & 3;
        if constexpr (MODE == 0) {
            _Float16* Y = (_Float16*)Yv;
            Y[(((size_t)b * NHEAD + hh) * SEQ + n0 + l16) * HDIM + d] = (_Float16)v;
        } else if constexpr (MODE == 1) {
            _Float16* Y = (_Float16*)Yv;
            Y[(((size_t)b * NHEAD + hh) * HDIM + d) * SEQ + n0 + l16] = (_Float16)v;
        } else {
            float* Y = (float*)Yv;
            Y[(size_t)b * FEAT * SEQ + (size_t)co * SEQ + n0 + l16] = v;
        }
    }
}

// ---------------------------------------------------------------------------
// Flash attention: 4 waves per block share one (b, head); each wave owns a
// 16-row n-tile (64 rows per block).  The m-loop streams 32-key chunks of
// K (4 KB, contiguous) and V (64 x 64B rows) into double-buffered LDS via
// the Tensor Data Mover (wave 0 issues; s_wait_tensorcnt pipelines one chunk
// ahead).  Softmax reductions are in-lane + one SWAPX16 swizzle.
// ---------------------------------------------------------------------------
__global__ void __launch_bounds__(128)
attn_flash(const _Float16* __restrict__ Qt,
           const _Float16* __restrict__ Kt,
           const _Float16* __restrict__ Vt,
           float* __restrict__ Ao)
{
    __shared__ __align__(128) _Float16 kbuf[2][32 * HDIM];  // [m][d]
    __shared__ __align__(128) _Float16 vbuf[2][HDIM * 32];  // [d][m]
    __shared__ __align__(32)  float    sc[4][16 * 32];
    __shared__ __align__(32)  float    cb[4][16];

    const int lane = threadIdx.x & 31;
    const int wid  = threadIdx.x >> 5;
    const int hi   = lane >> 4;
    const int l16  = lane & 15;

    const int bid   = blockIdx.x;             // b*128 + h*32 + ntile64
    const int nt64  = bid & 31;
    const int h     = (bid >> 5) & 3;
    const int b     = bid >> 7;
    const int n0    = nt64 * 64 + wid * 16;

    const size_t bh = (size_t)b * NHEAD + h;
    const _Float16* Qp = Qt + (bh * SEQ + n0) * HDIM;
    const _Float16* Kp = Kt + bh * SEQ * HDIM;
    const _Float16* Vp = Vt + bh * HDIM * SEQ;

    float* scw = sc[wid];
    float* cbw = cb[wid];

    // Q A-matrices (rows n, K-dim d): contiguous 32B run per lane (k = 16*hi + s)
    const _Float16* qrow = Qp + l16 * HDIM;
    const v16h aq0 = *(const v16h*)(qrow + 16 * hi);
    const v16h aq1 = *(const v16h*)(qrow + 32 + 16 * hi);

    v8f o0 = {}, o1 = {}, o2 = {}, o3 = {};
    float rowm = -3.0e38f, rowl = 0.0f;

#if USE_TDM
    const unsigned ka[2] = { lds_addr32(kbuf[0]), lds_addr32(kbuf[1]) };
    const unsigned va[2] = { lds_addr32(vbuf[0]), lds_addr32(vbuf[1]) };
    if (wid == 0) {                      // TDM ignores EXEC: one DMA per instr
        tdm_load_2d(Kp, ka[0], 512, 1, 512);          // 4 KB contiguous
        tdm_load_2d(Vp, va[0], 8, HDIM, SEQ / 4);     // 64 rows x 64B, 4 KB stride
    }
#else
    {
        const int t = threadIdx.x;
        const uint4* ks = (const uint4*)Kp;
        uint4* kd = (uint4*)kbuf[0];
        kd[t] = ks[t]; kd[t + 128] = ks[t + 128];
        const int row = t >> 1, seg = (t & 1) * 2;
        const uint4* vs = (const uint4*)(Vp + (size_t)row * SEQ);
        uint4* vd = (uint4*)(vbuf[0] + row * 32);
        vd[seg] = vs[seg]; vd[seg + 1] = vs[seg + 1];
    }
#endif

    #pragma unroll 1
    for (int it = 0; it < SEQ / 32; ++it) {
        const int mc  = it * 32;
        const int cur = it & 1;

#if USE_TDM
        if (wid == 0) {
            if (it + 1 < SEQ / 32) {
                tdm_load_2d(Kp + (size_t)(mc + 32) * HDIM, ka[cur ^ 1], 512, 1, 512);
                tdm_load_2d(Vp + (mc + 32),                va[cur ^ 1], 8, HDIM, SEQ / 4);
                __builtin_amdgcn_s_wait_tensorcnt(2);   // current chunk's 2 DMAs done
            } else {
                __builtin_amdgcn_s_wait_tensorcnt(0);
            }
        }
#else
        if (it + 1 < SEQ / 32) {
            const int t = threadIdx.x;
            const uint4* ks = (const uint4*)(Kp + (size_t)(mc + 32) * HDIM);
            uint4* kd = (uint4*)kbuf[cur ^ 1];
            kd[t] = ks[t]; kd[t + 128] = ks[t + 128];
            const int row = t >> 1, seg = (t & 1) * 2;
            const uint4* vs = (const uint4*)(Vp + (size_t)row * SEQ + mc + 32);
            uint4* vd = (uint4*)(vbuf[cur ^ 1] + row * 32);
            vd[seg] = vs[seg]; vd[seg + 1] = vs[seg + 1];
        }
#endif
        __syncthreads();                 // chunk `cur` visible to all waves

        const _Float16* kb = kbuf[cur];
        const _Float16* vb = vbuf[cur];

        // ---- scores: S = Q^T K, two 16x16 tiles, operands from LDS ----
        v8f s0 = {}, s1 = {};
        s0 = wmma_f16(aq0, *(const v16h*)(kb + (l16) * HDIM + 16 * hi), s0);
        s0 = wmma_f16(aq1, *(const v16h*)(kb + (l16) * HDIM + 32 + 16 * hi), s0);
        s1 = wmma_f16(aq0, *(const v16h*)(kb + (16 + l16) * HDIM + 16 * hi), s1);
        s1 = wmma_f16(aq1, *(const v16h*)(kb + (16 + l16) * HDIM + 32 + 16 * hi), s1);

        // ---- D-layout -> A-layout through per-wave LDS (scaled fp32) ----
        #pragma unroll
        for (int r = 0; r < 8; ++r) {
            scw[(r + 8 * hi) * 32 + l16]      = s0[r] * 0.125f;
            scw[(r + 8 * hi) * 32 + 16 + l16] = s1[r] * 0.125f;
        }
        __syncthreads();
        const v8f x0 = *(const v8f*)(scw + l16 * 32 + 16 * hi);
        const v8f x1 = *(const v8f*)(scw + l16 * 32 + 16 * hi + 8);

        // ---- online softmax: in-lane reduce + one SWAPX16 ----
        float mx = -3.0e38f;
        #pragma unroll
        for (int i = 0; i < 8; ++i) { mx = fmaxf(mx, x0[i]); mx = fmaxf(mx, x1[i]); }
        mx = fmaxf(mx, swap16(mx));
        const float mnew = fmaxf(rowm, mx);
        const float corr = __expf(rowm - mnew);
        rowm = mnew;

        float sum = 0.0f;
        v16h pa = {};                     // exp'd scores ARE the A-matrix
        #pragma unroll
        for (int i = 0; i < 8; ++i) {
            const float p0 = __expf(x0[i] - mnew);
            const float p1 = __expf(x1[i] - mnew);
            sum += p0 + p1;
            pa[i]     = (_Float16)p0;
            pa[8 + i] = (_Float16)p1;
        }
        sum += swap16(sum);
        rowl = rowl * corr + sum;

        // ---- distribute corr (row domain) to D-layout accumulators ----
        cbw[l16] = corr;
        __syncthreads();
        const v8f cv = *(const v8f*)(cbw + 8 * hi);
        o0 *= cv; o1 *= cv; o2 *= cv; o3 *= cv;

        // ---- O += P * V, B-operand from LDS (32B per lane) ----
        o0 = wmma_f16(pa, *(const v16h*)(vb + (l16) * 32 + 16 * hi), o0);
        o1 = wmma_f16(pa, *(const v16h*)(vb + (16 + l16) * 32 + 16 * hi), o1);
        o2 = wmma_f16(pa, *(const v16h*)(vb + (32 + l16) * 32 + 16 * hi), o2);
        o3 = wmma_f16(pa, *(const v16h*)(vb + (48 + l16) * 32 + 16 * hi), o3);
        __syncthreads();                 // all done with buf[cur] before reuse
    }

    // ---- normalize, store fp32 canonical [B,256,SEQ], c = 4d+h ----
    cbw[l16] = rowl;
    __syncthreads();
    const v8f lv = *(const v8f*)(cbw + 8 * hi);
    const size_t base = (size_t)b * FEAT * SEQ;
    #pragma unroll
    for (int r = 0; r < 8; ++r) {
        const float inv = 1.0f / lv[r];
        const int n = n0 + r + 8 * hi;
        Ao[base + (size_t)(4 * ( 0 + l16) + h) * SEQ + n] = o0[r] * inv;
        Ao[base + (size_t)(4 * (16 + l16) + h) * SEQ + n] = o1[r] * inv;
        Ao[base + (size_t)(4 * (32 + l16) + h) * SEQ + n] = o2[r] * inv;
        Ao[base + (size_t)(4 * (48 + l16) + h) * SEQ + n] = o3[r] * inv;
    }
}

// ---------------------------------------------------------------------------
extern "C" void kernel_launch(void* const* d_in, const int* in_sizes, int n_in,
                              void* d_out, int out_size, void* d_ws, size_t ws_size,
                              hipStream_t stream)
{
    (void)in_sizes; (void)n_in; (void)out_size; (void)ws_size;

    const float* q  = (const float*)d_in[0];
    const float* k  = (const float*)d_in[1];
    const float* v  = (const float*)d_in[2];
    const float* wq = (const float*)d_in[3];
    const float* bq = (const float*)d_in[4];
    const float* wk = (const float*)d_in[5];
    const float* bk = (const float*)d_in[6];
    const float* wv = (const float*)d_in[7];
    const float* bv = (const float*)d_in[8];
    const float* wm = (const float*)d_in[9];
    const float* bm = (const float*)d_in[10];

    const size_t elems = (size_t)BATCH * FEAT * SEQ;             // 2M
    _Float16* Qt = (_Float16*)d_ws;                              // 4 MB
    _Float16* Kt = Qt + elems;                                   // 4 MB
    _Float16* Vt = Kt + elems;                                   // 4 MB
    float*    Ah = (float*)((char*)d_ws + 12ull * 1024 * 1024);  // 8 MB

    const dim3 gg(BATCH * 16 * 16), gt(256);
    gemm256<0><<<gg, gt, 0, stream>>>(wq, bq, q, Qt);
    gemm256<0><<<gg, gt, 0, stream>>>(wk, bk, k, Kt);
    gemm256<1><<<gg, gt, 0, stream>>>(wv, bv, v, Vt);

    attn_flash<<<dim3(BATCH * NHEAD * (SEQ / 64)), dim3(128), 0, stream>>>(Qt, Kt, Vt, Ah);

    gemm256<2><<<gg, gt, 0, stream>>>(wm, bm, Ah, (void*)d_out);
}
